// Llama4MegaBlocksRouter_1460288880658
// MI455X (gfx1250) — compile-verified, matrix operand order checked by reading, not run on hardware
//
#include <hip/hip_runtime.h>

// Llama4 MoE router on MI455X (gfx1250, wave32, WMMA + TDM).
// logits = x[32768,5120] @ W[128,5120]^T, top-1 + sigmoid + scatter.
// Split-bf16 (hi/lo) 3-product WMMA GEMM for ~fp32 accuracy at bf16 matrix rate.
// W is pre-split into bf16 hi/lo planes once (L2-resident), then DMA'd per-chunk
// into LDS by the Tensor Data Mover while the VALU converts the x chunk.

#define HIDDEN 5120
#define NEXP   128
#define TOKENS 32768
#define BT     32     // tokens per block (2 x 16-token tiles per wave)
#define BK     64     // K chunk staged in LDS per iteration
#define WELEMS (NEXP * HIDDEN)   // 655360 elements per plane

typedef __attribute__((ext_vector_type(16))) __bf16 bf16x16;
typedef __attribute__((ext_vector_type(8)))  __bf16 bf16x8;
typedef __attribute__((ext_vector_type(8)))  float  f32x8;
typedef __attribute__((ext_vector_type(4)))  unsigned u32x4;
typedef __attribute__((ext_vector_type(8)))  int      i32x8;

union V16 { bf16x16 v; bf16x8 h[2]; };

__device__ __forceinline__ unsigned f32_to_bf16_rne(float f) {
    unsigned u = __float_as_uint(f);
    unsigned r = u + 0x7FFFu + ((u >> 16) & 1u);
    return r >> 16;
}
__device__ __forceinline__ float bf16_bits_to_f32(unsigned h) {
    return __uint_as_float(h << 16);
}

// float4 -> packed bf16 hi quad + lo quad
__device__ __forceinline__ void split4(float4 v, uint2* hi_out, uint2* lo_out) {
    float f[4] = {v.x, v.y, v.z, v.w};
    unsigned hi[4], lo[4];
#pragma unroll
    for (int i = 0; i < 4; ++i) {
        hi[i] = f32_to_bf16_rne(f[i]);
        float resid = f[i] - bf16_bits_to_f32(hi[i]);
        lo[i] = f32_to_bf16_rne(resid);
    }
    *hi_out = make_uint2(hi[0] | (hi[1] << 16), hi[2] | (hi[3] << 16));
    *lo_out = make_uint2(lo[0] | (lo[1] << 16), lo[2] | (lo[3] << 16));
}

#define WMMA_BF16(A, B, C) \
    __builtin_amdgcn_wmma_f32_16x16x32_bf16(false, (A), false, (B), (short)0, (C), false, false)

// ---------------------------------------------------------------------------
// One-time: split W f32 -> bf16 hi plane + lo plane (contiguous) in d_ws.
// ---------------------------------------------------------------------------
__global__ __launch_bounds__(256) void wsplit_kernel(
    const float* __restrict__ W,
    unsigned short* __restrict__ whi,
    unsigned short* __restrict__ wlo)
{
    int i = (blockIdx.x * 256 + threadIdx.x) * 4;
    float4 v = *(const float4*)(W + i);
    uint2 hi, lo;
    split4(v, &hi, &lo);
    *(uint2*)(whi + i) = hi;
    *(uint2*)(wlo + i) = lo;
}

// ---------------------------------------------------------------------------
// Main router kernel.
// ---------------------------------------------------------------------------
__global__ __launch_bounds__(256) void router_topk_kernel(
    const float* __restrict__ x,
    const unsigned short* __restrict__ wbf,   // [2][NEXP][HIDDEN] bf16 (hi, lo planes)
    float* __restrict__ scores,               // [TOKENS, NEXP]
    float* __restrict__ exp_w,                // [TOKENS]
    int*   __restrict__ top_idx)              // [TOKENS]
{
    __shared__ unsigned short xa_hi[BT][BK];       // 4 KB
    __shared__ unsigned short xa_lo[BT][BK];       // 4 KB
    __shared__ unsigned short wb[2][NEXP][BK];     // 32 KB (hi plane, lo plane) - TDM target
    __shared__ float logits_s[BT][NEXP];           // 16 KB
    __shared__ float sig_s[BT];
    __shared__ int   idx_s[BT];

    const int tid  = threadIdx.x;
    const int wid  = tid >> 5;        // wave id 0..7 -> expert strip [16*wid, 16*wid+16)
    const int lane = tid & 31;
    const int l15  = lane & 15;
    const int ksel = lane >> 4;       // lane group selects K sub-block per ISA layout
    const int tok0 = blockIdx.x * BT;

    f32x8 c0 = {};  // tokens [tok0..tok0+15]   x 16 experts
    f32x8 c1 = {};  // tokens [tok0+16..tok0+31] x 16 experts

    const int col4 = tid & 15;        // float4 column within BK
    const int srow = tid >> 4;        // staging row group

    // Wave-uniform pieces of the Tensor DMA Descriptor (D#), forced into SGPRs.
    const unsigned lds_base =
        __builtin_amdgcn_readfirstlane((unsigned)(size_t)(void*)&wb[0][0][0]);
    const unsigned wlo32 = __builtin_amdgcn_readfirstlane((unsigned)(size_t)wbf);
    const unsigned whi32 =
        __builtin_amdgcn_readfirstlane((unsigned)(((size_t)wbf) >> 32));

    for (int k0 = 0; k0 < HIDDEN; k0 += BK) {
        __syncthreads();  // previous iteration's LDS readers done; wb free for DMA

        // ---- Wave 0: kick TDM for the W chunk: 3D tile X=64, Y=128, Z=2 planes ----
        if (wid == 0) {
            unsigned long long gaddr =
                (((unsigned long long)whi32 << 32) | wlo32) +
                (unsigned long long)__builtin_amdgcn_readfirstlane(k0) * 2ull;
            u32x4 g0 = { 1u,                               // count=1, user mode
                         lds_base,                         // D#.lds_addr
                         (unsigned)(gaddr & 0xFFFFFFFFull),
                         (unsigned)((gaddr >> 32) & 0x1FFFFFFull) | (2u << 30) }; // type=2
            i32x8 g1 = { (int)(1u << 16),                  // data_size=2B, mask=0, no pad
                         (int)((unsigned)HIDDEN << 16),    // tensor_dim0[15:0] @bit48
                         (int)((unsigned)NEXP << 16),      // tensor_dim1[15:0] @bit80
                         (int)((unsigned)BK << 16),        // tile_dim0 @bit112
                         (int)((unsigned)NEXP | (2u << 16)), // tile_dim1=128, tile_dim2=2
                         (int)HIDDEN,                      // tensor_dim0_stride (48b) lo32
                         0,                                // stride0 hi / dim1_stride lo16 (0)
                         (int)(((unsigned)HIDDEN * NEXP) >> 16) }; // dim1_stride bits 47:16
            u32x4 g2 = { 2u,                               // tensor_dim2 = 2 planes
                         0u,                               // tensor_dim3 = 0
                         (unsigned)(HIDDEN * NEXP),        // tensor_dim2_stride lo32
                         0u };                             // stride hi / tile_dim3=0
            u32x4 g3 = { 0u, 0u, 0u, 0u };
            asm volatile("tensor_load_to_lds %0, %1, %2, %3"
                         :: "s"(g0), "s"(g1), "s"(g2), "s"(g3)
                         : "memory");
        }

        // ---- All threads: stage x chunk (32x64 f32 -> bf16 hi/lo), overlaps DMA ----
#pragma unroll
        for (int p = 0; p < 2; ++p) {
            int row = p * 16 + srow;
            float4 v = ((const float4*)(x + (size_t)(tok0 + row) * HIDDEN + k0))[col4];
            uint2 hi, lo;
            split4(v, &hi, &lo);
            *(uint2*)&xa_hi[row][col4 * 4] = hi;
            *(uint2*)&xa_lo[row][col4 * 4] = lo;
        }

        if (wid == 0) __builtin_amdgcn_s_wait_tensorcnt(0);
        __syncthreads();

        // Two K=32 WMMA steps per chunk. Operand layout per CDNA5 ISA:
        // 16-bit A/B 16x32: lanes 0-15 hold K 0-7 & 16-23, lanes 16-31 hold K 8-15 & 24-31.
#pragma unroll
        for (int kk = 0; kk < BK; kk += 32) {
            const int o0 = kk + ksel * 8;
            const int o1 = kk + 16 + ksel * 8;
            const int brow = wid * 16 + l15;

            V16 bh, bl, ah0, al0, ah1, al1;
            bh.h[0]  = *(const bf16x8*)&wb[0][brow][o0];
            bh.h[1]  = *(const bf16x8*)&wb[0][brow][o1];
            bl.h[0]  = *(const bf16x8*)&wb[1][brow][o0];
            bl.h[1]  = *(const bf16x8*)&wb[1][brow][o1];
            ah0.h[0] = *(const bf16x8*)&xa_hi[l15][o0];
            ah0.h[1] = *(const bf16x8*)&xa_hi[l15][o1];
            al0.h[0] = *(const bf16x8*)&xa_lo[l15][o0];
            al0.h[1] = *(const bf16x8*)&xa_lo[l15][o1];
            ah1.h[0] = *(const bf16x8*)&xa_hi[16 + l15][o0];
            ah1.h[1] = *(const bf16x8*)&xa_hi[16 + l15][o1];
            al1.h[0] = *(const bf16x8*)&xa_lo[16 + l15][o0];
            al1.h[1] = *(const bf16x8*)&xa_lo[16 + l15][o1];

            // Split product: hi*hi + lo*hi + hi*lo (lo*lo term ~2^-32, dropped)
            c0 = WMMA_BF16(ah0.v, bh.v, c0);
            c0 = WMMA_BF16(al0.v, bh.v, c0);
            c0 = WMMA_BF16(ah0.v, bl.v, c0);
            c1 = WMMA_BF16(ah1.v, bh.v, c1);
            c1 = WMMA_BF16(al1.v, bh.v, c1);
            c1 = WMMA_BF16(ah1.v, bl.v, c1);
        }
    }

    // C/D layout: VGPR r, lanes 0-15 -> (M=r, N=lane); lanes 16-31 -> (M=8+r, N=lane-16)
    {
        const int e = wid * 16 + l15;
#pragma unroll
        for (int r = 0; r < 8; ++r) {
            logits_s[r + 8 * ksel][e]      = c0[r];
            logits_s[16 + r + 8 * ksel][e] = c1[r];
        }
    }
    __syncthreads();

    // Per-token top-1 over 128 experts; strict '>' keeps lowest index on ties (jax top_k).
    if (tid < BT) {
        float best = logits_s[tid][0];
        int bi = 0;
#pragma unroll 4
        for (int e = 1; e < NEXP; ++e) {
            float v = logits_s[tid][e];
            if (v > best) { best = v; bi = e; }
        }
        float sg = 1.0f / (1.0f + __expf(-best));
        sig_s[tid] = sg;
        idx_s[tid] = bi;
        exp_w[tok0 + tid]   = sg;
        top_idx[tok0 + tid] = bi;
    }
    __syncthreads();

    // Scatter: sigmoid(top) at argmax, 0 elsewhere (sigmoid(-inf)=0). 16 floats/thread.
    {
        const int t     = tid >> 3;
        const int ebase = (tid & 7) * 16;
        const int bi    = idx_s[t];
        const float sg  = sig_s[t];
        float4* outp = (float4*)(scores + (size_t)(tok0 + t) * NEXP + ebase);
#pragma unroll
        for (int q = 0; q < 4; ++q) {
            float4 v;
            v.x = (ebase + q * 4 + 0 == bi) ? sg : 0.0f;
            v.y = (ebase + q * 4 + 1 == bi) ? sg : 0.0f;
            v.z = (ebase + q * 4 + 2 == bi) ? sg : 0.0f;
            v.w = (ebase + q * 4 + 3 == bi) ? sg : 0.0f;
            outp[q] = v;
        }
    }
}

extern "C" void kernel_launch(void* const* d_in, const int* in_sizes, int n_in,
                              void* d_out, int out_size, void* d_ws, size_t ws_size,
                              hipStream_t stream) {
    (void)in_sizes; (void)n_in; (void)out_size; (void)ws_size;
    const float* x = (const float*)d_in[0];
    const float* W = (const float*)d_in[1];

    unsigned short* whi = (unsigned short*)d_ws;       // [NEXP*HIDDEN] bf16 hi plane
    unsigned short* wlo = whi + WELEMS;                // [NEXP*HIDDEN] bf16 lo plane

    float* scores = (float*)d_out;                                   // [T, E]
    float* exp_w  = scores + (size_t)TOKENS * NEXP;                  // [T]
    int*   tidx   = (int*)(scores + (size_t)TOKENS * NEXP + TOKENS); // [T] int32

    // One-time split of W into bf16 hi/lo planes (re-run each launch: deterministic).
    wsplit_kernel<<<WELEMS / (256 * 4), 256, 0, stream>>>(W, whi, wlo);

    router_topk_kernel<<<TOKENS / BT, 256, 0, stream>>>(x, whi, scores, exp_w, tidx);
}
